// Model_87651692577195
// MI455X (gfx1250) — matrix operand order repaired and checked by reading, not compiled, not run on hardware
//
#include <hip/hip_runtime.h>

#define USER_N 50000
#define ITEM_N 50000
#define TOT_N  (USER_N + ITEM_N)
#define LATD   128
#define BM     128
#define BK     32
#define APAD   (BK + 2)   // 34 elems -> 68B rows, gcd(17 dwords, 64 banks)=1

typedef __attribute__((ext_vector_type(16))) __bf16 v16bf;
typedef __attribute__((ext_vector_type(8)))  float  v8f;

__device__ __forceinline__ __bf16 f2bf(float f) {
  unsigned u = __builtin_bit_cast(unsigned, f);
  u += 0x7FFFu + ((u >> 16) & 1u);                 // round-to-nearest-even
  unsigned short h = (unsigned short)(u >> 16);
  return __builtin_bit_cast(__bf16, h);
}

// ---------------------------------------------------------------------------
// GEMM: out[M x 128] (+)= w * LeakyReLU_0.2( A[M x K] @ W[K x 128] )
// w = softmax(modal_weight)[widx].  bf16 WMMA 16x16x32, f32 accumulate.
// 256 threads = 8 waves; wave w owns rows [16w,16w+16) of a 128x128 tile.
// ---------------------------------------------------------------------------
__global__ __launch_bounds__(256)
void k_gemm_lrelu_bf16(const float* __restrict__ A, const float* __restrict__ W,
                       float* __restrict__ out, int M, int K,
                       const float* __restrict__ mw, int widx, int accumulate) {
  __shared__ __bf16 As[BM][APAD];     // A tile, row-major [m][k]
  __shared__ __bf16 Bs[LATD][APAD];   // W tile, transposed [n][k]

  const int tid  = threadIdx.x;
  const int wave = tid >> 5;
  const int lane = tid & 31;
  const int half = lane >> 4;
  const int r16  = lane & 15;
  const long rowBase = (long)blockIdx.x * BM;

  v8f acc[8];
#pragma unroll
  for (int i = 0; i < 8; ++i)
#pragma unroll
    for (int e = 0; e < 8; ++e) acc[i][e] = 0.0f;

  const int kChunks = K / BK;
  for (int kc = 0; kc < kChunks; ++kc) {
    const int k0 = kc * BK;
    // Stage A tile (128x32): 4096 elems, 16 per thread, coalesced.
#pragma unroll
    for (int i = 0; i < 16; ++i) {
      int idx = tid + i * 256;
      int r = idx >> 5, c = idx & 31;
      long gr = rowBase + r;
      float v = (gr < M) ? A[gr * (long)K + k0 + c] : 0.0f;
      As[r][c] = f2bf(v);
      if (i == 0 && kc + 1 < kChunks)
        __builtin_prefetch(&A[gr * (long)K + k0 + BK + c], 0, 1); // global_prefetch_b8
    }
    // Stage W tile (32x128) transposed into Bs[n][k].
#pragma unroll
    for (int i = 0; i < 16; ++i) {
      int idx = tid + i * 256;
      int kr = idx >> 7, nc = idx & 127;
      Bs[nc][kr] = f2bf(W[(long)(k0 + kr) * LATD + nc]);
    }
    __syncthreads();

    // A fragment: 16x32 bf16 per ISA layout (half-wave selects K groups).
    v16bf a;
#pragma unroll
    for (int j = 0; j < 8; ++j) {
      a[j]     = As[wave * 16 + r16][half * 8 + j];
      a[8 + j] = As[wave * 16 + r16][16 + half * 8 + j];
    }
#pragma unroll
    for (int nb = 0; nb < 8; ++nb) {
      v16bf b;
#pragma unroll
      for (int j = 0; j < 16; ++j)
        b[j] = Bs[nb * 16 + r16][(j < 8) ? (half * 8 + j) : (16 + half * 8 + (j - 8))];
      acc[nb] = __builtin_amdgcn_wmma_f32_16x16x32_bf16(
          false, a, false, b, (short)0, acc[nb], false, false);
    }
    __syncthreads();
  }

  // Epilogue: softmax weight, LeakyReLU(0.2), optional accumulate.
  float e0 = expf(mw[0]), e1 = expf(mw[1]);
  float w = (widx == 0 ? e0 : e1) / (e0 + e1);
#pragma unroll
  for (int nb = 0; nb < 8; ++nb) {
#pragma unroll
    for (int v = 0; v < 8; ++v) {
      long row = rowBase + wave * 16 + v + 8 * half;
      int col = nb * 16 + r16;
      if (row < M) {
        float x = acc[nb][v];
        x = x > 0.0f ? x : 0.2f * x;
        float val = w * x;
        float* p = out + row * LATD + col;
        if (accumulate) *p += val; else *p = val;
      }
    }
  }
}

// ---------------------------------------------------------------------------
// SpMM: out[r] += alpha * val * x[c], one wave32 per edge, float4 per lane.
// Edge index forced scalar (readfirstlane) so rows/cols/vals become s_load_b32
// through the constant cache instead of 32 redundant per-lane vmem loads.
// Gathers/scatters stay L2-resident (x and out are 51.2MB each, L2 = 192MB).
// ---------------------------------------------------------------------------
__global__ __launch_bounds__(256)
void k_spmm_atomic(const int* __restrict__ rows, const int* __restrict__ cols,
                   const float* __restrict__ vals, const float* __restrict__ x,
                   float* __restrict__ out, float alpha, int E) {
  int ew = (int)((blockIdx.x * (long)blockDim.x + threadIdx.x) >> 5);
  int lane = threadIdx.x & 31;
  if (ew >= E) return;
  int e = __builtin_amdgcn_readfirstlane(ew);   // wave-uniform edge id
  int r = rows[e], c = cols[e];
  float v = alpha * vals[e];
  const float4 xv = *reinterpret_cast<const float4*>(x + (size_t)c * LATD + lane * 4);
  float* o = out + (size_t)r * LATD + lane * 4;
  __hip_atomic_fetch_add(o + 0, v * xv.x, __ATOMIC_RELAXED, __HIP_MEMORY_SCOPE_AGENT);
  __hip_atomic_fetch_add(o + 1, v * xv.y, __ATOMIC_RELAXED, __HIP_MEMORY_SCOPE_AGENT);
  __hip_atomic_fetch_add(o + 2, v * xv.z, __ATOMIC_RELAXED, __HIP_MEMORY_SCOPE_AGENT);
  __hip_atomic_fetch_add(o + 3, v * xv.w, __ATOMIC_RELAXED, __HIP_MEMORY_SCOPE_AGENT);
}

// ---------------------------------------------------------------------------
// Row L2-normalize in place: one wave per 128-float row.
// ---------------------------------------------------------------------------
__global__ __launch_bounds__(256)
void k_l2norm_rows(float* __restrict__ x, int rows) {
  int w = (int)((blockIdx.x * (long)blockDim.x + threadIdx.x) >> 5);
  int lane = threadIdx.x & 31;
  if (w >= rows) return;
  float4* p = reinterpret_cast<float4*>(x + (size_t)w * LATD) + lane;
  float4 v = *p;
  float s = v.x * v.x + v.y * v.y + v.z * v.z + v.w * v.w;
#pragma unroll
  for (int off = 16; off > 0; off >>= 1) s += __shfl_xor(s, off, 32);
  float inv = 1.0f / fmaxf(sqrtf(s), 1e-12f);
  v.x *= inv; v.y *= inv; v.z *= inv; v.w *= inv;
  *p = v;
}

// dst = src + 0.5 * l2norm_row(src)   (initializes GNN accumulator)
__global__ __launch_bounds__(256)
void k_final_init(float* __restrict__ dst, const float* __restrict__ src, int rows) {
  int w = (int)((blockIdx.x * (long)blockDim.x + threadIdx.x) >> 5);
  int lane = threadIdx.x & 31;
  if (w >= rows) return;
  const float4 v = *(reinterpret_cast<const float4*>(src + (size_t)w * LATD) + lane);
  float s = v.x * v.x + v.y * v.y + v.z * v.z + v.w * v.w;
#pragma unroll
  for (int off = 16; off > 0; off >>= 1) s += __shfl_xor(s, off, 32);
  float h = 0.5f / fmaxf(sqrtf(s), 1e-12f);
  float4 o;
  o.x = v.x + h * v.x; o.y = v.y + h * v.y; o.z = v.z + h * v.z; o.w = v.w + h * v.w;
  *(reinterpret_cast<float4*>(dst + (size_t)w * LATD) + lane) = o;
}

// ---------------------------------------------------------------------------
// Elementwise helpers (float4 granularity; all sizes here are /4 counts).
// ---------------------------------------------------------------------------
__global__ void k_zero4(float4* p, long n4) {
  long i = blockIdx.x * (long)blockDim.x + threadIdx.x;
  if (i < n4) p[i] = make_float4(0.f, 0.f, 0.f, 0.f);
}
__global__ void k_copy4(float4* __restrict__ d, const float4* __restrict__ s, long n4) {
  long i = blockIdx.x * (long)blockDim.x + threadIdx.x;
  if (i < n4) d[i] = s[i];
}
__global__ void k_add4(float4* __restrict__ d, const float4* __restrict__ s, long n4) {
  long i = blockIdx.x * (long)blockDim.x + threadIdx.x;
  if (i < n4) {
    float4 a = d[i], b = s[i];
    a.x += b.x; a.y += b.y; a.z += b.z; a.w += b.w;
    d[i] = a;
  }
}
__global__ void k_avg4(float4* __restrict__ d, const float4* __restrict__ a,
                       const float4* __restrict__ b, long n4) {
  long i = blockIdx.x * (long)blockDim.x + threadIdx.x;
  if (i < n4) {
    float4 x = a[i], y = b[i];
    x.x = 0.5f * (x.x + y.x); x.y = 0.5f * (x.y + y.y);
    x.z = 0.5f * (x.z + y.z); x.w = 0.5f * (x.w + y.w);
    d[i] = x;
  }
}

static inline long cdiv(long a, long b) { return (a + b - 1) / b; }

extern "C" void kernel_launch(void* const* d_in, const int* in_sizes, int n_in,
                              void* d_out, int out_size, void* d_ws, size_t ws_size,
                              hipStream_t stream) {
  const int*   adj_rows  = (const int*)d_in[0];
  const int*   adj_cols  = (const int*)d_in[1];
  const float* adj_vals  = (const float*)d_in[2];
  const int*   id_rows   = (const int*)d_in[3];
  const int*   id_cols   = (const int*)d_in[4];
  const float* id_vals   = (const float*)d_in[5];
  const int*   feat_rows = (const int*)d_in[6];
  const int*   feat_cols = (const int*)d_in[7];
  const float* feat_vals = (const float*)d_in[8];
  const float* uEmbeds   = (const float*)d_in[9];
  const float* iEmbeds   = (const float*)d_in[10];
  const float* img_emb   = (const float*)d_in[11];
  const float* img_id    = (const float*)d_in[12];
  const float* txt_emb   = (const float*)d_in[13];
  const float* txt_id    = (const float*)d_in[14];
  const float* img_tr    = (const float*)d_in[15];
  const float* img_id_tr = (const float*)d_in[16];
  const float* txt_tr    = (const float*)d_in[17];
  const float* txt_id_tr = (const float*)d_in[18];
  const float* mw        = (const float*)d_in[19];
  const int E = in_sizes[0];

  const long SZ  = (long)TOT_N * LATD;          // 12.8M floats per buffer
  const long U4  = (long)USER_N * LATD / 4;
  const long I4  = (long)ITEM_N * LATD / 4;
  const long N4  = SZ / 4;

  float* b0 = (float*)d_ws;          // concat(u, i)
  float* b1 = b0 + SZ;               // xId / scratch
  float* b2 = b1 + SZ;               // xFeat / scratch
  float* b4 = b2 + SZ;               // embedsId
  float* b6 = b4 + SZ;               // embedsFeat
  float* b7 = b6 + SZ;               // GNN tmp
  float* b8 = b7 + SZ;               // embedsModal
  float* outp = (float*)d_out;

  dim3 blk(256);
  const long gemmBlocks = cdiv(ITEM_N, BM);
  const long spmmBlocks = cdiv((long)E, 8);         // 8 waves (edges) per block
  const long rowWaveBlk = cdiv((long)TOT_N * 32, 256);
  const long rowWaveItm = cdiv((long)ITEM_N * 32, 256);

#define EW(n4) dim3((unsigned)cdiv((n4), 256))

  // ---- Phase 1: modal GEMMs (bf16 WMMA) into item halves of b1 / b2 ----
  float* allID   = b1 + (long)USER_N * LATD;
  float* allFeat = b2 + (long)USER_N * LATD;
  hipLaunchKernelGGL(k_gemm_lrelu_bf16, dim3((unsigned)gemmBlocks), blk, 0, stream,
                     img_id, img_id_tr, allID, ITEM_N, 1024, mw, 0, 0);
  hipLaunchKernelGGL(k_gemm_lrelu_bf16, dim3((unsigned)gemmBlocks), blk, 0, stream,
                     txt_id, txt_id_tr, allID, ITEM_N, 768, mw, 1, 1);
  hipLaunchKernelGGL(k_gemm_lrelu_bf16, dim3((unsigned)gemmBlocks), blk, 0, stream,
                     img_emb, img_tr, allFeat, ITEM_N, 1024, mw, 0, 0);
  hipLaunchKernelGGL(k_gemm_lrelu_bf16, dim3((unsigned)gemmBlocks), blk, 0, stream,
                     txt_emb, txt_tr, allFeat, ITEM_N, 768, mw, 1, 1);
  hipLaunchKernelGGL(k_l2norm_rows, dim3((unsigned)rowWaveItm), blk, 0, stream, allID, ITEM_N);
  hipLaunchKernelGGL(k_l2norm_rows, dim3((unsigned)rowWaveItm), blk, 0, stream, allFeat, ITEM_N);

  // ---- Build concat inputs ----
  hipLaunchKernelGGL(k_copy4, EW(U4), blk, 0, stream, (float4*)b0, (const float4*)uEmbeds, U4);
  hipLaunchKernelGGL(k_copy4, EW(I4), blk, 0, stream, (float4*)(b0 + (long)USER_N * LATD),
                     (const float4*)iEmbeds, I4);
  hipLaunchKernelGGL(k_copy4, EW(U4), blk, 0, stream, (float4*)b1, (const float4*)uEmbeds, U4);
  hipLaunchKernelGGL(k_copy4, EW(U4), blk, 0, stream, (float4*)b2, (const float4*)uEmbeds, U4);

  // ---- ID branch: b4 = spmm(adj,xId) + spmm(adj, concat(b4[:U], i)) + 0.2*spmm(id, ui) ----
  hipLaunchKernelGGL(k_zero4, EW(N4), blk, 0, stream, (float4*)b4, N4);
  hipLaunchKernelGGL(k_spmm_atomic, dim3((unsigned)spmmBlocks), blk, 0, stream,
                     adj_rows, adj_cols, adj_vals, b1, b4, 1.0f, E);
  hipLaunchKernelGGL(k_copy4, EW(U4), blk, 0, stream, (float4*)b1, (const float4*)b4, U4);
  hipLaunchKernelGGL(k_copy4, EW(I4), blk, 0, stream, (float4*)(b1 + (long)USER_N * LATD),
                     (const float4*)iEmbeds, I4);
  hipLaunchKernelGGL(k_spmm_atomic, dim3((unsigned)spmmBlocks), blk, 0, stream,
                     adj_rows, adj_cols, adj_vals, b1, b4, 1.0f, E);
  hipLaunchKernelGGL(k_spmm_atomic, dim3((unsigned)spmmBlocks), blk, 0, stream,
                     id_rows, id_cols, id_vals, b0, b4, 0.2f, E);

  // ---- Feature branch: same shape into b6 ----
  hipLaunchKernelGGL(k_zero4, EW(N4), blk, 0, stream, (float4*)b6, N4);
  hipLaunchKernelGGL(k_spmm_atomic, dim3((unsigned)spmmBlocks), blk, 0, stream,
                     adj_rows, adj_cols, adj_vals, b2, b6, 1.0f, E);
  hipLaunchKernelGGL(k_copy4, EW(U4), blk, 0, stream, (float4*)b2, (const float4*)b6, U4);
  hipLaunchKernelGGL(k_copy4, EW(I4), blk, 0, stream, (float4*)(b2 + (long)USER_N * LATD),
                     (const float4*)iEmbeds, I4);
  hipLaunchKernelGGL(k_spmm_atomic, dim3((unsigned)spmmBlocks), blk, 0, stream,
                     adj_rows, adj_cols, adj_vals, b2, b6, 1.0f, E);
  hipLaunchKernelGGL(k_spmm_atomic, dim3((unsigned)spmmBlocks), blk, 0, stream,
                     feat_rows, feat_cols, feat_vals, b0, b6, 0.2f, E);

  // ---- embedsModal = 0.5*(b4 + b6);  out = modal + 0.5*l2norm(modal) ----
  hipLaunchKernelGGL(k_avg4, EW(N4), blk, 0, stream, (float4*)b8, (const float4*)b4,
                     (const float4*)b6, N4);
  hipLaunchKernelGGL(k_final_init, dim3((unsigned)rowWaveBlk), blk, 0, stream, outp, b8, TOT_N);

  // ---- GNN layer 1: b7 = spmm(adj, modal); out += b7 ----
  hipLaunchKernelGGL(k_zero4, EW(N4), blk, 0, stream, (float4*)b7, N4);
  hipLaunchKernelGGL(k_spmm_atomic, dim3((unsigned)spmmBlocks), blk, 0, stream,
                     adj_rows, adj_cols, adj_vals, b8, b7, 1.0f, E);
  hipLaunchKernelGGL(k_add4, EW(N4), blk, 0, stream, (float4*)outp, (const float4*)b7, N4);

  // ---- GNN layer 2: b1 = spmm(adj, b7); out += b1 ----
  hipLaunchKernelGGL(k_zero4, EW(N4), blk, 0, stream, (float4*)b1, N4);
  hipLaunchKernelGGL(k_spmm_atomic, dim3((unsigned)spmmBlocks), blk, 0, stream,
                     adj_rows, adj_cols, adj_vals, b7, b1, 1.0f, E);
  hipLaunchKernelGGL(k_add4, EW(N4), blk, 0, stream, (float4*)outp, (const float4*)b1, N4);
#undef EW
}
